// Encoder_11141145166302
// MI455X (gfx1250) — matrix operand order repaired and checked by reading, not compiled, not run on hardware
//
#include <hip/hip_runtime.h>
#include <hip/hip_bf16.h>

// ---------------- problem constants ----------------
#define T_STEPS 128
#define BATCH   512
#define DIN     300
#define DPAD    320   // DIN padded up to a multiple of 32 (WMMA K granularity)
#define HID     512
#define G3H     1536  // 3*HID
#define NLAYER  3

typedef __attribute__((ext_vector_type(16))) __bf16 v16bf;
typedef __attribute__((ext_vector_type(8)))  __bf16 v8bf;
typedef __attribute__((ext_vector_type(8)))  float  v8f;

// ---------------- fp32 -> bf16 (with K-padding) ----------------
__global__ void cvt_pad_bf16(const float* __restrict__ src, __bf16* __restrict__ dst,
                             int rows, int incols, int outcols) {
    long idx = (long)blockIdx.x * blockDim.x + threadIdx.x;
    long total = (long)rows * outcols;
    if (idx >= total) return;
    int  c = (int)(idx % outcols);
    long r = idx / outcols;
    float v = (c < incols) ? src[r * (long)incols + c] : 0.0f;
    dst[idx] = (__bf16)v;
}

// ---------------- batched WMMA GEMM ----------------
// C[M,N] = A[M,K] * W[N,K]^T (+ bias[N]); A,W bf16 row-major (lda=ldw=K), C fp32.
// Block: 128 threads = 4 waves (2 down x 2 across), block tile 64(M) x 128(N).
// Wave tile: 32(M) x 64(N) = 2x4 WMMA tiles -> 8 wmma per 12 b128-loads per K-chunk.
struct GemmJob {
    const __bf16* A;
    const __bf16* W;
    const float*  bias;   // nullptr -> no bias
    float*        C;
    int M, N, K;
};
struct GemmBatch { GemmJob job[4]; };

__global__ __launch_bounds__(128)
void gemm_wmma(GemmBatch batch, int njobs) {
    int jz = blockIdx.z;
    if (jz >= njobs) return;
    GemmJob j = batch.job[jz];

    int m_blk = blockIdx.y * 64;
    int n_blk = blockIdx.x * 128;
    if (m_blk >= j.M || n_blk >= j.N) return;

    int wave = threadIdx.x >> 5;      // 0..3: 2 down (M) x 2 across (N)
    int lane = threadIdx.x & 31;
    int row0 = m_blk + (wave & 1) * 32;
    int col0 = n_blk + (wave >> 1) * 64;
    int lhalf = lane >> 4;            // 0 or 1
    int lmod  = lane & 15;

    v8f acc[2][4] = {};
    const int K = j.K;

    for (int k0 = 0; k0 < K; k0 += 32) {
        // A fragments (contiguous K order, matching B):
        // lane<16 -> K {k0..k0+15}; lane>=16 -> +16
        v16bf aF[2];
#pragma unroll
        for (int i = 0; i < 2; ++i) {
            const __bf16* ap = j.A + (size_t)(row0 + i * 16 + lmod) * K + k0 + lhalf * 16;
            union { v16bf v; v8bf h[2]; } u;
            u.h[0] = *(const v8bf*)ap;
            u.h[1] = *(const v8bf*)(ap + 8);
            aF[i] = u.v;
        }
        // B fragments: lane holds column n = col0+tile*16+lmod,
        // 16 contiguous K starting at k0 + lhalf*16 (== 16 contiguous elems of W row n)
        v16bf bF[4];
#pragma unroll
        for (int i = 0; i < 4; ++i) {
            const __bf16* wp = j.W + (size_t)(col0 + i * 16 + lmod) * K + k0 + lhalf * 16;
            bF[i] = *(const v16bf*)wp;
        }
#pragma unroll
        for (int a = 0; a < 2; ++a)
#pragma unroll
            for (int b = 0; b < 4; ++b)
                acc[a][b] = __builtin_amdgcn_wmma_f32_16x16x32_bf16(
                    false, aF[a], false, bF[b], (short)0, acc[a][b], false, false);
    }

    // C/D layout: vgpr r -> M = (lane/16)*8 + r, N = lane%16
#pragma unroll
    for (int a = 0; a < 2; ++a) {
#pragma unroll
        for (int b = 0; b < 4; ++b) {
            int ccol = col0 + b * 16 + lmod;
            float bv = j.bias ? j.bias[ccol] : 0.0f;
#pragma unroll
            for (int r = 0; r < 8; ++r) {
                int crow = row0 + a * 16 + lhalf * 8 + r;
                j.C[(size_t)crow * j.N + ccol] = acc[a][b][r] + bv;
            }
        }
    }
}

// ---------------- GRU gate elementwise update (both directions via grid.z) ----------------
struct GruPtrs {
    const float* gi;   // [B, 3H]
    const float* gh;   // [B, 3H]
    const float* bih;  // [3H]
    const float* bhh;  // [3H]
    float*  h;         // [B, H]  in/out (fp32 state)
    __bf16* hbf;       // [B, H]  out (bf16 copy for next GEMM)
};

__global__ void gru_update(GruPtrs p0, GruPtrs p1) {
    GruPtrs p = blockIdx.z ? p1 : p0;
    int idx  = blockIdx.x * blockDim.x + threadIdx.x;   // 0 .. B*H-1
    int j    = idx & (HID - 1);
    int b    = idx >> 9;                                // /HID
    const float* gi = p.gi + (size_t)b * G3H;
    const float* gh = p.gh + (size_t)b * G3H;

    float r = gi[j]           + p.bih[j]           + gh[j]           + p.bhh[j];
    r = 1.0f / (1.0f + __expf(-r));
    float z = gi[HID + j]     + p.bih[HID + j]     + gh[HID + j]     + p.bhh[HID + j];
    z = 1.0f / (1.0f + __expf(-z));
    float n = gi[2 * HID + j] + p.bih[2 * HID + j] + r * (gh[2 * HID + j] + p.bhh[2 * HID + j]);
    n = tanhf(n);

    float hprev = p.h[idx];
    float hn = (1.0f - z) * n + z * hprev;
    p.h[idx]   = hn;
    p.hbf[idx] = (__bf16)hn;
}

// ---------------- concat fw/bw final hidden -> bf16 [B, 2H] ----------------
__global__ void concat_bf16(const float* __restrict__ hf, const float* __restrict__ hb,
                            __bf16* __restrict__ out) {
    int idx = blockIdx.x * blockDim.x + threadIdx.x;    // 0 .. B*2H-1
    int c = idx & (2 * HID - 1);
    int b = idx >> 10;
    float v = (c < HID) ? hf[b * HID + c] : hb[b * HID + (c - HID)];
    out[idx] = (__bf16)v;
}

// ---------------- host orchestration ----------------
extern "C" void kernel_launch(void* const* d_in, const int* in_sizes, int n_in,
                              void* d_out, int out_size, void* d_ws, size_t ws_size,
                              hipStream_t stream) {
    (void)in_sizes; (void)n_in; (void)out_size; (void)ws_size;

    const float* X      = (const float*)d_in[0];
    const float* Wih0f  = (const float*)d_in[2];
    const float* Whh0f  = (const float*)d_in[3];
    const float* Bih0f  = (const float*)d_in[4];
    const float* Bhh0f  = (const float*)d_in[5];
    const float* WihLf  = (const float*)d_in[6];   // [2,3H,H]
    const float* WhhLf  = (const float*)d_in[7];
    const float* BihLf  = (const float*)d_in[8];   // [2,3H]
    const float* BhhLf  = (const float*)d_in[9];
    const float* Wih0b  = (const float*)d_in[10];
    const float* Whh0b_ = (const float*)d_in[11];
    const float* Bih0b  = (const float*)d_in[12];
    const float* Bhh0b  = (const float*)d_in[13];
    const float* WihLb  = (const float*)d_in[14];
    const float* WhhLb  = (const float*)d_in[15];
    const float* BihLb  = (const float*)d_in[16];
    const float* BhhLb  = (const float*)d_in[17];
    const float* Wmu    = (const float*)d_in[18];  // [H, 2H]
    const float* bmu    = (const float*)d_in[19];
    const float* Wlv    = (const float*)d_in[20];
    const float* blv    = (const float*)d_in[21];

    // ---- carve workspace ----
    char* base = (char*)d_ws;
    size_t off = 0;
    auto alloc = [&](size_t bytes) -> char* {
        char* p = base + off;
        off = (off + bytes + 255) & ~(size_t)255;
        return p;
    };
    __bf16* Xb        = (__bf16*)alloc((size_t)T_STEPS * BATCH * DPAD * 2);
    __bf16* Wih0bf[2] = { (__bf16*)alloc((size_t)G3H * DPAD * 2),
                          (__bf16*)alloc((size_t)G3H * DPAD * 2) };
    __bf16* Whh0bf[2] = { (__bf16*)alloc((size_t)G3H * HID * 2),
                          (__bf16*)alloc((size_t)G3H * HID * 2) };
    __bf16* Wihbf[2][2], *Whhbf[2][2];
    for (int d = 0; d < 2; ++d)
        for (int l = 0; l < 2; ++l) {
            Wihbf[d][l] = (__bf16*)alloc((size_t)G3H * HID * 2);
            Whhbf[d][l] = (__bf16*)alloc((size_t)G3H * HID * 2);
        }
    __bf16* Wmubf = (__bf16*)alloc((size_t)HID * 2 * HID * 2);
    __bf16* Wlvbf = (__bf16*)alloc((size_t)HID * 2 * HID * 2);

    float*  H32 = (float*)alloc((size_t)2 * NLAYER * BATCH * HID * 4);
    __bf16* Hbf = (__bf16*)alloc((size_t)2 * NLAYER * BATCH * HID * 2);
    float*  GI  = (float*)alloc((size_t)2 * BATCH * G3H * 4);
    float*  GH  = (float*)alloc((size_t)2 * BATCH * G3H * 4);
    __bf16* SEMB = (__bf16*)alloc((size_t)BATCH * 2 * HID * 2);

    auto h32 = [&](int d, int l) { return H32 + (size_t)(d * NLAYER + l) * BATCH * HID; };
    auto hbf = [&](int d, int l) { return Hbf + (size_t)(d * NLAYER + l) * BATCH * HID; };
    auto gip = [&](int d) { return GI + (size_t)d * BATCH * G3H; };
    auto ghp = [&](int d) { return GH + (size_t)d * BATCH * G3H; };

    // ---- one-time conversions (captured each call; deterministic) ----
    auto cvt = [&](const float* src, __bf16* dst, long rows, int incols, int outcols) {
        long total = rows * (long)outcols;
        int blocks = (int)((total + 255) / 256);
        cvt_pad_bf16<<<dim3(blocks), dim3(256), 0, stream>>>(src, dst, (int)rows, incols, outcols);
    };
    cvt(X,      Xb,        (long)T_STEPS * BATCH, DIN, DPAD);
    cvt(Wih0f,  Wih0bf[0], G3H, DIN, DPAD);
    cvt(Wih0b,  Wih0bf[1], G3H, DIN, DPAD);
    cvt(Whh0f,  Whh0bf[0], G3H, HID, HID);
    cvt(Whh0b_, Whh0bf[1], G3H, HID, HID);
    for (int l = 0; l < 2; ++l) {
        cvt(WihLf + (size_t)l * G3H * HID, Wihbf[0][l], G3H, HID, HID);
        cvt(WhhLf + (size_t)l * G3H * HID, Whhbf[0][l], G3H, HID, HID);
        cvt(WihLb + (size_t)l * G3H * HID, Wihbf[1][l], G3H, HID, HID);
        cvt(WhhLb + (size_t)l * G3H * HID, Whhbf[1][l], G3H, HID, HID);
    }
    cvt(Wmu, Wmubf, HID, 2 * HID, 2 * HID);
    cvt(Wlv, Wlvbf, HID, 2 * HID, 2 * HID);

    hipMemsetAsync(H32, 0, (size_t)2 * NLAYER * BATCH * HID * 4, stream);
    hipMemsetAsync(Hbf, 0, (size_t)2 * NLAYER * BATCH * HID * 2, stream);

    // ---- recurrence: per (t, layer) one batched GEMM (fw/bw x gi/gh) + one gate kernel ----
    dim3 ggrid(G3H / 128, BATCH / 64, 4);         // 12 x 8 x 4 = 384 blocks
    dim3 egrid(BATCH * HID / 256, 1, 2);

    for (int t = 0; t < T_STEPS; ++t) {
        for (int l = 0; l < NLAYER; ++l) {
            GemmBatch gb{};
            for (int d = 0; d < 2; ++d) {
                GemmJob gi_j, gh_j;
                if (l == 0) {
                    int ti = (d == 0) ? t : (T_STEPS - 1 - t);
                    gi_j = GemmJob{ Xb + (size_t)ti * BATCH * DPAD, Wih0bf[d], nullptr,
                                    gip(d), BATCH, G3H, DPAD };
                    gh_j = GemmJob{ hbf(d, 0), Whh0bf[d], nullptr,
                                    ghp(d), BATCH, G3H, HID };
                } else {
                    gi_j = GemmJob{ hbf(d, l - 1), Wihbf[d][l - 1], nullptr,
                                    gip(d), BATCH, G3H, HID };
                    gh_j = GemmJob{ hbf(d, l), Whhbf[d][l - 1], nullptr,
                                    ghp(d), BATCH, G3H, HID };
                }
                gb.job[2 * d]     = gi_j;
                gb.job[2 * d + 1] = gh_j;
            }
            gemm_wmma<<<ggrid, 128, 0, stream>>>(gb, 4);

            const float* bih_f = (l == 0) ? Bih0f : BihLf + (size_t)(l - 1) * G3H;
            const float* bhh_f = (l == 0) ? Bhh0f : BhhLf + (size_t)(l - 1) * G3H;
            const float* bih_b = (l == 0) ? Bih0b : BihLb + (size_t)(l - 1) * G3H;
            const float* bhh_b = (l == 0) ? Bhh0b : BhhLb + (size_t)(l - 1) * G3H;
            GruPtrs p0{ gip(0), ghp(0), bih_f, bhh_f, h32(0, l), hbf(0, l) };
            GruPtrs p1{ gip(1), ghp(1), bih_b, bhh_b, h32(1, l), hbf(1, l) };
            gru_update<<<egrid, 256, 0, stream>>>(p0, p1);
        }
    }

    // ---- heads: concat(h_fw[2], h_bw[2]) -> mu / logvar into d_out ----
    concat_bf16<<<dim3((BATCH * 2 * HID) / 256), dim3(256), 0, stream>>>(h32(0, 2), h32(1, 2), SEMB);

    float* out = (float*)d_out;
    GemmBatch fb{};
    fb.job[0] = GemmJob{ SEMB, Wmubf, bmu, out,                BATCH, HID, 2 * HID };
    fb.job[1] = GemmJob{ SEMB, Wlvbf, blv, out + BATCH * HID,  BATCH, HID, 2 * HID };
    gemm_wmma<<<dim3(HID / 128, BATCH / 64, 2), 128, 0, stream>>>(fb, 2);
}
